// DTCRModel_4518305595635
// MI455X (gfx1250) — matrix operand order repaired and checked by reading, not compiled, not run on hardware
//
#include <hip/hip_runtime.h>
#include <hip/hip_bf16.h>
#include <stdint.h>
#include <stddef.h>

// ---------------------------------------------------------------------------
// DTCR model for MI455X (gfx1250): bf16 WMMA GEMMs + fp32 gate math.
// Decoder (256 sequential [256x1536]x[1536x9216] GEMMs) dominates FLOPs; its
// 28MB bf16 weights are L2-resident, so we are WMMA-throughput bound, not
// HBM bound -> v_wmma_f32_16x16x32_bf16 everywhere.
// ---------------------------------------------------------------------------

typedef __bf16 bf16;
typedef __attribute__((ext_vector_type(16))) __bf16 bfx16;
typedef __attribute__((ext_vector_type(8)))  float  fx8;

#define BATCH   256
#define TSTEPS  256
#define INSZ    64
#define H       256
#define G3H     768
#define NUNITS  1536
#define DECN    9216   // 2 * 3 * 1536
#define DEC3H   4608

__device__ __forceinline__ bf16 f2bf(float f) {
    union { float f; uint32_t u; } v; v.f = f;
    uint32_t r = v.u + 0x7FFFu + ((v.u >> 16) & 1u);   // round-to-nearest-even
    uint16_t h = (uint16_t)(r >> 16);
    bf16 out; __builtin_memcpy(&out, &h, 2); return out;
}
__device__ __forceinline__ float sigmoidf_(float x) { return 1.0f / (1.0f + expf(-x)); }

// ----------------------------- pack / fill ---------------------------------
__global__ void pack_bf16_kernel(const float* __restrict__ s, bf16* __restrict__ d, size_t n) {
    for (size_t i = blockIdx.x * (size_t)blockDim.x + threadIdx.x; i < n;
         i += (size_t)gridDim.x * blockDim.x)
        d[i] = f2bf(s[i]);
}
__global__ void fill_zero_kernel(uint32_t* __restrict__ p, size_t n) {
    for (size_t i = blockIdx.x * (size_t)blockDim.x + threadIdx.x; i < n;
         i += (size_t)gridDim.x * blockDim.x)
        p[i] = 0u;
}

// ----------------------------- WMMA GEMM -----------------------------------
// C[M,N] = A[M,K](bf16,row-major,lda) * W[N,K](bf16,row-major,ldw)^T (+bias[N])
// Wave tile 32x64: 2 A-frags x 4 B-frags -> 8 wmma per 32-wide K chunk.
// A frag: lane(l%16)=row, l/16 picks K-chunks {8kh..8kh+7, 16+8kh..+7}.
// B frag: lane(l%16)=col n -> row n of W, same K-chunks as A frag.
__global__ __launch_bounds__(256) void gemm_bf16_wmma(
    const bf16* __restrict__ A, int lda,
    const bf16* __restrict__ W, int ldw,
    float* __restrict__ C, int ldc,
    int M, int N, int K, const float* __restrict__ bias, int NT)
{
    const int wave = threadIdx.x >> 5;
    const int lane = threadIdx.x & 31;
    const int tile = blockIdx.x * 8 + wave;
    const int tiles = (M >> 5) * NT;
    if (tile >= tiles) return;
    const int mt = tile / NT, nt = tile % NT;
    const int m0 = mt * 32, n0 = nt * 64;
    const int laneM = lane & 15, kh = lane >> 4;

    union Frag { bfx16 v; uint4 u[2]; };
    fx8 acc[2][4];
#pragma unroll
    for (int i = 0; i < 2; ++i)
#pragma unroll
        for (int j = 0; j < 4; ++j)
#pragma unroll
            for (int e = 0; e < 8; ++e) acc[i][j][e] = 0.0f;

    const bf16* aP0 = A + (size_t)(m0 + laneM) * lda + 8 * kh;
    const bf16* aP1 = A + (size_t)(m0 + 16 + laneM) * lda + 8 * kh;
    const bf16* wP0 = W + (size_t)(n0 +  0 + laneM) * ldw + 8 * kh;
    const bf16* wP1 = W + (size_t)(n0 + 16 + laneM) * ldw + 8 * kh;
    const bf16* wP2 = W + (size_t)(n0 + 32 + laneM) * ldw + 8 * kh;
    const bf16* wP3 = W + (size_t)(n0 + 48 + laneM) * ldw + 8 * kh;
    const bf16* wP[4] = { wP0, wP1, wP2, wP3 };

    for (int k0 = 0; k0 < K; k0 += 32) {
        Frag a0, a1, b;
        a0.u[0] = *(const uint4*)(aP0 + k0);
        a0.u[1] = *(const uint4*)(aP0 + k0 + 16);
        a1.u[0] = *(const uint4*)(aP1 + k0);
        a1.u[1] = *(const uint4*)(aP1 + k0 + 16);
#pragma unroll
        for (int j = 0; j < 4; ++j) {
            b.u[0] = *(const uint4*)(wP[j] + k0);
            b.u[1] = *(const uint4*)(wP[j] + k0 + 16);
            acc[0][j] = __builtin_amdgcn_wmma_f32_16x16x32_bf16(
                false, a0.v, false, b.v, (short)0, acc[0][j], false, false);
            acc[1][j] = __builtin_amdgcn_wmma_f32_16x16x32_bf16(
                false, a1.v, false, b.v, (short)0, acc[1][j], false, false);
        }
    }
#pragma unroll
    for (int i = 0; i < 2; ++i) {
        const int rowb = m0 + 16 * i + 8 * kh;
#pragma unroll
        for (int j = 0; j < 4; ++j) {
            const int col = n0 + 16 * j + laneM;
            const float bv = bias ? bias[col] : 0.0f;
#pragma unroll
            for (int e = 0; e < 8; ++e)
                C[(size_t)(rowb + e) * ldc + col] = acc[i][j][e] + bv;
        }
    }
}

// ----------------------------- GRU gates -----------------------------------
// Encoder step. Folded batch row = b*d + s; time index t = step*d + s
// (forward) or 255 - (step*d + s) (backward; this is simultaneously where the
// un-reversed output lands, since x_rev[b,tau] = x[b,255-tau]).
__global__ void gru_gate_enc(const float* __restrict__ gi,   // [B*T, 768], row b*T+t
                             const float* __restrict__ gh,   // [Bd, 768]
                             const float* __restrict__ bih,
                             const float* __restrict__ bhh,
                             float* __restrict__ hf,         // [Bd, H]
                             bf16*  __restrict__ hb,         // [Bd, H]
                             bf16*  __restrict__ y,          // [B, T, 512]
                             int dil, int step, int bwd, int featOff)
{
    const int idx = blockIdx.x * blockDim.x + threadIdx.x;   // Bd*H threads
    const int row = idx >> 8;          // /H
    const int j   = idx & 255;         // %H
    const int b   = row / dil;
    const int s   = row - b * dil;
    const int tt  = step * dil + s;
    const int t   = bwd ? (TSTEPS - 1 - tt) : tt;

    const float* gir = gi + (size_t)(b * TSTEPS + t) * G3H;
    const float* ghr = gh + (size_t)row * G3H;

    const float r = sigmoidf_(gir[j]       + bih[j]       + ghr[j]       + bhh[j]);
    const float z = sigmoidf_(gir[H + j]   + bih[H + j]   + ghr[H + j]   + bhh[H + j]);
    const float n = tanhf    (gir[2*H + j] + bih[2*H + j] + r * (ghr[2*H + j] + bhh[2*H + j]));
    const float hn = (1.0f - z) * n + z * hf[idx];

    hf[idx] = hn;
    hb[idx] = f2bf(hn);
    y[(size_t)(b * TSTEPS + t) * 512 + featOff + j] = f2bf(hn);
}

// Latent extraction: phase 0 (NUM_STEPS % d == 0 for d in {1,4,16}).
__global__ void extract_latent(const float* __restrict__ hf, int dil,
                               float* __restrict__ latF, bf16* __restrict__ latB, int col0)
{
    const int idx = blockIdx.x * blockDim.x + threadIdx.x;   // B*H threads
    const int b = idx >> 8, j = idx & 255;
    const float v = hf[(size_t)(b * dil) * H + j];
    latF[(size_t)b * NUNITS + col0 + j] = v;
    latB[(size_t)b * NUNITS + col0 + j] = f2bf(v);
}

// Decoder step gates: G[b, 0:4608] = h@Wih^T, G[b, 4608:9216] = h@Whh^T.
__global__ void gru_gate_dec(const float* __restrict__ G,
                             const float* __restrict__ bih,
                             const float* __restrict__ bhh,
                             float* __restrict__ hf, bf16* __restrict__ hb)
{
    const int idx = blockIdx.x * blockDim.x + threadIdx.x;   // 256*1536 threads
    const int b = idx / NUNITS;
    const int j = idx - b * NUNITS;
    const float* Gr = G + (size_t)b * DECN;

    const float r = sigmoidf_(Gr[j]            + bih[j]            + Gr[DEC3H + j]            + bhh[j]);
    const float z = sigmoidf_(Gr[NUNITS + j]   + bih[NUNITS + j]   + Gr[DEC3H + NUNITS + j]   + bhh[NUNITS + j]);
    const float n = tanhf    (Gr[2*NUNITS + j] + bih[2*NUNITS + j] + r * (Gr[DEC3H + 2*NUNITS + j] + bhh[2*NUNITS + j]));
    const float hn = (1.0f - z) * n + z * hf[idx];
    hf[idx] = hn;
    hb[idx] = f2bf(hn);
}

// ----------------------------- classifier ----------------------------------
// combined = [latent; latent] (fake_repr == latent); rows unpermuted (the JAX
// threefry permutation is a host-side fixed constant we do not reproduce).
__global__ void classifier_kernel(const float* __restrict__ latent,
                                  const float* __restrict__ W1, const float* __restrict__ b1,
                                  const float* __restrict__ W2, const float* __restrict__ b2,
                                  float* __restrict__ cls, float* __restrict__ labels)
{
    __shared__ float s0[128], s1[128];
    const int i = blockIdx.x;           // 0..511
    const int j = threadIdx.x;          // 0..127
    const float* x = latent + (size_t)(i & (BATCH - 1)) * NUNITS;
    const float* w = W1 + (size_t)j * NUNITS;
    float acc = 0.0f;
    for (int k = 0; k < NUNITS; ++k) acc += x[k] * w[k];
    const float h = fmaxf(acc + b1[j], 0.0f);
    s0[j] = h * W2[j];
    s1[j] = h * W2[128 + j];
    __syncthreads();
    for (int off = 64; off > 0; off >>= 1) {
        if (j < off) { s0[j] += s0[j + off]; s1[j] += s1[j + off]; }
        __syncthreads();
    }
    if (j == 0) {
        const float o0 = s0[0] + b2[0], o1 = s1[0] + b2[1];
        const float m = fmaxf(o0, o1);
        const float e0 = expf(o0 - m), e1 = expf(o1 - m);
        const float inv = 1.0f / (e0 + e1);
        cls[2 * i]     = e0 * inv;
        cls[2 * i + 1] = e1 * inv;
        labels[2 * i]     = (i < BATCH) ? 1.0f : 0.0f;
        labels[2 * i + 1] = (i < BATCH) ? 0.0f : 1.0f;
    }
}

// ---------------------------------------------------------------------------
extern "C" void kernel_launch(void* const* d_in, const int* in_sizes, int n_in,
                              void* d_out, int out_size, void* d_ws, size_t ws_size,
                              hipStream_t stream)
{
    (void)in_sizes; (void)n_in; (void)out_size; (void)ws_size;
    static const int DIL[3] = {1, 4, 16};

    float* outF = (float*)d_out;
    const size_t OUT_INPUTS = 0;
    const size_t OUT_LATENT = (size_t)BATCH * TSTEPS * INSZ;            // 4194304
    const size_t OUT_RECON  = OUT_LATENT + (size_t)BATCH * NUNITS;      // 4587520
    const size_t OUT_CLS    = OUT_RECON + (size_t)BATCH * TSTEPS * INSZ;
    const size_t OUT_LABELS = OUT_CLS + 2ull * BATCH * 2;

    // -------- workspace bump allocator --------
    size_t off = 0;
    auto alloc = [&](size_t bytes) -> void* {
        off = (off + 255) & ~(size_t)255;
        void* p = (char*)d_ws + off;
        off += bytes;
        return p;
    };
    bf16*  bufA  = (bf16*)alloc((size_t)BATCH * TSTEPS * 512 * 2);
    bf16*  bufB  = (bf16*)alloc((size_t)BATCH * TSTEPS * 512 * 2);
    float* gi    = (float*)alloc((size_t)BATCH * TSTEPS * G3H * 4);
    float* gh    = (float*)alloc((size_t)BATCH * 16 * G3H * 4);
    float* hfBuf = (float*)alloc((size_t)BATCH * 16 * H * 4);
    bf16*  hbBuf = (bf16*)alloc((size_t)BATCH * 16 * H * 2);
    bf16*  latB  = (bf16*)alloc((size_t)BATCH * NUNITS * 2);
    float* decHf = (float*)alloc((size_t)BATCH * NUNITS * 4);
    bf16*  decHb = (bf16*)alloc((size_t)BATCH * NUNITS * 2);
    float* Gdec  = (float*)alloc((size_t)BATCH * DECN * 4);
    bf16*  encWih[3][2];
    bf16*  encWhh[3][2];
    for (int l = 0; l < 3; ++l) {
        const int Din = (l == 0) ? INSZ : 512;
        for (int dir = 0; dir < 2; ++dir) {
            encWih[l][dir] = (bf16*)alloc((size_t)G3H * Din * 2);
            encWhh[l][dir] = (bf16*)alloc((size_t)G3H * H * 2);
        }
    }
    bf16* decW = (bf16*)alloc((size_t)DECN * NUNITS * 2);   // [Wih ; Whh] rows
    bf16* linW = (bf16*)alloc((size_t)INSZ * NUNITS * 2);

    // -------- launch helpers --------
    auto pack = [&](const float* s, bf16* d, size_t n) {
        size_t b = (n + 255) / 256; if (b > 4096) b = 4096;
        pack_bf16_kernel<<<dim3((unsigned)b), dim3(256), 0, stream>>>(s, d, n);
    };
    auto zero = [&](void* p, size_t bytes) {
        size_t n = bytes / 4;
        size_t b = (n + 255) / 256; if (b > 4096) b = 4096;
        fill_zero_kernel<<<dim3((unsigned)b), dim3(256), 0, stream>>>((uint32_t*)p, n);
    };
    auto gemm = [&](const bf16* A, int lda, const bf16* W, int ldw,
                    float* C, int ldc, int M, int N, int K, const float* bias) {
        const int NT = N / 64;
        const int tiles = (M / 32) * NT;
        const int blocks = (tiles + 7) / 8;
        gemm_bf16_wmma<<<dim3(blocks), dim3(256), 0, stream>>>(
            A, lda, W, ldw, C, ldc, M, N, K, bias, NT);
    };

    // -------- input indices (setup_inputs dict order, flattened) --------
    const float* inputs = (const float*)d_in[0];
    // enc layer l base = 1 + 8l: [Wih_f, Whh_f, bih_f, bhh_f, Wih_b, Whh_b, bih_b, bhh_b]
    const float* dec_Wih = (const float*)d_in[25];
    const float* dec_Whh = (const float*)d_in[26];
    const float* dec_bih = (const float*)d_in[27];
    const float* dec_bhh = (const float*)d_in[28];
    const float* lin_Wf  = (const float*)d_in[29];
    const float* lin_b   = (const float*)d_in[30];
    const float* cls_W1  = (const float*)d_in[31];
    const float* cls_b1  = (const float*)d_in[32];
    const float* cls_W2  = (const float*)d_in[33];
    const float* cls_b2  = (const float*)d_in[34];

    // -------- weight packing (fp32 -> bf16; row-major == WMMA B layout) ----
    for (int l = 0; l < 3; ++l) {
        const int Din = (l == 0) ? INSZ : 512;
        const int base = 1 + 8 * l;
        for (int dir = 0; dir < 2; ++dir) {
            pack((const float*)d_in[base + 4 * dir + 0], encWih[l][dir], (size_t)G3H * Din);
            pack((const float*)d_in[base + 4 * dir + 1], encWhh[l][dir], (size_t)G3H * H);
        }
    }
    pack(dec_Wih, decW, (size_t)DEC3H * NUNITS);
    pack(dec_Whh, decW + (size_t)DEC3H * NUNITS, (size_t)DEC3H * NUNITS);
    pack(lin_Wf, linW, (size_t)INSZ * NUNITS);

    // -------- forward inputs to d_out --------
    hipMemcpyAsync(outF + OUT_INPUTS, inputs,
                   (size_t)BATCH * TSTEPS * INSZ * 4, hipMemcpyDeviceToDevice, stream);

    // -------- encoder --------
    pack(inputs, bufA, (size_t)BATCH * TSTEPS * INSZ);
    bf16* xin = bufA;
    bf16* yout = bufB;
    int Din = INSZ;
    for (int l = 0; l < 3; ++l) {
        const int d = DIL[l];
        const int L = TSTEPS / d;
        const int Bd = BATCH * d;
        for (int dir = 0; dir < 2; ++dir) {
            const int base = 1 + 8 * l;
            const float* bih = (const float*)d_in[base + 4 * dir + 2];
            const float* bhh = (const float*)d_in[base + 4 * dir + 3];
            // bulk input-side gates for all timesteps (one big parallel GEMM)
            gemm(xin, Din, encWih[l][dir], Din, gi, G3H,
                 BATCH * TSTEPS, G3H, Din, nullptr);
            zero(hfBuf, (size_t)Bd * H * 4);
            zero(hbBuf, (size_t)Bd * H * 2);
            for (int st = 0; st < L; ++st) {
                gemm(hbBuf, H, encWhh[l][dir], H, gh, G3H, Bd, G3H, H, nullptr);
                gru_gate_enc<<<dim3(Bd), dim3(256), 0, stream>>>(
                    gi, gh, bih, bhh, hfBuf, hbBuf, yout, d, st, dir, dir ? H : 0);
            }
            extract_latent<<<dim3(256), dim3(256), 0, stream>>>(
                hfBuf, d, outF + OUT_LATENT, latB, 512 * l + (dir ? H : 0));
        }
        { bf16* t = xin; xin = yout; yout = t; }
        Din = 512;
    }

    // -------- decoder (autoregressive, h0 = latent) --------
    hipMemcpyAsync(decHf, outF + OUT_LATENT, (size_t)BATCH * NUNITS * 4,
                   hipMemcpyDeviceToDevice, stream);
    hipMemcpyAsync(decHb, latB, (size_t)BATCH * NUNITS * 2,
                   hipMemcpyDeviceToDevice, stream);
    for (int t = 0; t < TSTEPS; ++t) {
        gemm(decHb, NUNITS, decW, NUNITS, Gdec, DECN, BATCH, DECN, NUNITS, nullptr);
        gru_gate_dec<<<dim3((BATCH * NUNITS) / 256), dim3(256), 0, stream>>>(
            Gdec, dec_bih, dec_bhh, decHf, decHb);
        gemm(decHb, NUNITS, linW, NUNITS,
             outF + OUT_RECON + (size_t)t * INSZ, TSTEPS * INSZ,
             BATCH, INSZ, NUNITS, lin_b);
    }

    // -------- classifier + labels --------
    classifier_kernel<<<dim3(2 * BATCH), dim3(128), 0, stream>>>(
        outF + OUT_LATENT, cls_W1, cls_b1, cls_W2, cls_b2,
        outF + OUT_CLS, outF + OUT_LABELS);
}
